// CrossRPEAttentionMulti_9869834846450
// MI455X (gfx1250) — matrix operand order repaired and checked by reading, not compiled, not run on hardware
//
#include <hip/hip_runtime.h>
#include <hip/hip_bf16.h>
#include <stdint.h>

// ---- static problem configuration (matches reference) ----
#define Bb   4
#define Cc   1024
#define Hh   16
#define HD   64
#define Pp   576
#define LQ   577          // queries (cls + 576 patches)
#define Nn   1729         // key/value tokens (3*576 + 1)
#define NP   1760         // N padded to multiple of 32 (55 key blocks)
#define LQP  592          // LQ padded to multiple of 16 (37 query tiles)
#define SCALE 0.125f      // HD^-0.5
#define BSTRIDE (Pp + 2)  // LDS bias-slab row stride (banking)

typedef __bf16 bf16;
typedef __attribute__((ext_vector_type(16))) __bf16 v16bf;
typedef __attribute__((ext_vector_type(8)))  float  v8f;

union FragU { v16bf v; uint4 q[2]; };

// A-fragment style: two contiguous 8-element (16B) runs
static __device__ __forceinline__ v16bf frag2x8(const bf16* p0, const bf16* p1) {
  FragU f;
  f.q[0] = *(const uint4*)p0;
  f.q[1] = *(const uint4*)p1;
  return f.v;
}
// B-fragment style: one contiguous 16-element (32B) run
static __device__ __forceinline__ v16bf frag16c(const bf16* p) {
  FragU f;
  f.q[0] = *(const uint4*)(p);
  f.q[1] = *(const uint4*)(p + 8);
  return f.v;
}
static __device__ __forceinline__ v8f wmma_bf16(v16bf a, v16bf b, v8f c) {
  // D = A(16x32) * B(32x16) + C, fp32 accumulate
  return __builtin_amdgcn_wmma_f32_16x16x32_bf16(false, a, false, b, (short)0, c,
                                                 false, false);
}

// ---------------- fp32 -> bf16 conversion (grid-stride) ----------------
__global__ void cvt_kernel(const float* __restrict__ src, bf16* __restrict__ dst, int n) {
  for (int i = blockIdx.x * blockDim.x + threadIdx.x; i < n; i += gridDim.x * blockDim.x)
    dst[i] = (bf16)src[i];
}

// ------------- q = q_learned + pos_embed, bf16, rows padded -------------
__global__ void qprep_kernel(const float* __restrict__ ql, const float* __restrict__ pe,
                             bf16* __restrict__ qbf) {
  const int total = LQP * Cc;
  for (int idx = blockIdx.x * blockDim.x + threadIdx.x; idx < total;
       idx += gridDim.x * blockDim.x) {
    int i = idx >> 10;                 // row (0..591)
    int c = idx & (Cc - 1);
    int ic = i < LQ ? i : (LQ - 1);    // clamp padded rows to valid data
    qbf[idx] = (bf16)(ql[c] + pe[ic * Cc + c]);
  }
}

// ------- rpe_tab[h][i][t] = sum_d q[i, h*64+d] * rpe_W[d][t]  (tiny) -------
__global__ void rpe_kernel(const float* __restrict__ ql, const float* __restrict__ pe,
                           const float* __restrict__ rpeW, float* __restrict__ rtab,
                           int nb) {
  const int total = Hh * LQ * nb;
  for (int idx = blockIdx.x * blockDim.x + threadIdx.x; idx < total;
       idx += gridDim.x * blockDim.x) {
    int t = idx % nb;
    int r = idx / nb;
    int i = r % LQ;
    int h = r / LQ;
    float s = 0.f;
    const int cb = h * HD;
    #pragma unroll 8
    for (int d = 0; d < HD; ++d)
      s += (ql[cb + d] + pe[i * Cc + cb + d]) * rpeW[d * nb + t];
    rtab[((h * LQ + i) << 3) + t] = s;   // stride padded to 8
  }
}

// ---------------- K / V projection GEMM (bf16 WMMA, double-buffered) ----------------
// grid = (NP/16, B*H, 2); one wave computes a 16-token x 64-dim (one head) tile.
// z==0: K[bh][tok][d] row-major ; z==1: Vt[bh][d][tok] transposed.
__global__ __launch_bounds__(32) void kv_gemm_kernel(
    const bf16* __restrict__ xbf, const bf16* __restrict__ wkbf,
    const bf16* __restrict__ wvbf, bf16* __restrict__ kout, bf16* __restrict__ vtout) {
  const int lane = threadIdx.x, ln = lane & 15, lh = lane >> 4;
  const int ntile = blockIdx.x;
  const int bh = blockIdx.y, b = bh >> 4, h = bh & 15;
  const int isv = blockIdx.z;
  const bf16* W = isv ? wvbf : wkbf;

  int tok = ntile * 16 + ln;
  if (tok > Nn - 1) tok = Nn - 1;                 // clamp padded-token reads
  const bf16* xrow = xbf + (size_t)(b * Nn + tok) * Cc;

  auto loadA = [&](int ks) -> v16bf {
    const bf16* ap = xrow + ks * 32 + lh * 16;
    return frag2x8(ap, ap + 8);
  };
  auto loadB = [&](int ks, int ct) -> v16bf {
    return frag16c(W + (size_t)(h * HD + ct * 16 + ln) * Cc + (size_t)ks * 32 + lh * 16);
  };

  v8f zero = {};
  v8f acc[4] = {zero, zero, zero, zero};

  // software pipeline: loads for step ks+1 are in flight during step ks's WMMAs
  v16bf a_c  = loadA(0);
  v16bf b_c0 = loadB(0, 0), b_c1 = loadB(0, 1), b_c2 = loadB(0, 2), b_c3 = loadB(0, 3);
  for (int ks = 0; ks < Cc / 32 - 1; ++ks) {
    v16bf a_n  = loadA(ks + 1);
    v16bf b_n0 = loadB(ks + 1, 0), b_n1 = loadB(ks + 1, 1);
    v16bf b_n2 = loadB(ks + 1, 2), b_n3 = loadB(ks + 1, 3);
    acc[0] = wmma_bf16(a_c, b_c0, acc[0]);
    acc[1] = wmma_bf16(a_c, b_c1, acc[1]);
    acc[2] = wmma_bf16(a_c, b_c2, acc[2]);
    acc[3] = wmma_bf16(a_c, b_c3, acc[3]);
    a_c = a_n; b_c0 = b_n0; b_c1 = b_n1; b_c2 = b_n2; b_c3 = b_n3;
  }
  acc[0] = wmma_bf16(a_c, b_c0, acc[0]);
  acc[1] = wmma_bf16(a_c, b_c1, acc[1]);
  acc[2] = wmma_bf16(a_c, b_c2, acc[2]);
  acc[3] = wmma_bf16(a_c, b_c3, acc[3]);

  #pragma unroll
  for (int ct = 0; ct < 4; ++ct) {
    #pragma unroll
    for (int p = 0; p < 8; ++p) {
      int trow = ntile * 16 + p + 8 * lh;          // < NP
      int d = ct * 16 + ln;
      bf16 v = (bf16)acc[ct][p];
      if (isv) vtout[((size_t)bh * HD + d) * NP + trow] = v;
      else     kout[((size_t)bh * NP + trow) * HD + d] = v;
    }
  }
}

// ---------------- fused flash attention with RPE bias ----------------
// grid = (LQP/16, B*H); one wave owns 16 queries, streams 55 key blocks of 32.
// Bias slab for the 16 queries is pre-gathered to LDS; K fragments are
// double-buffered across blocks; V fragments load early to overlap softmax VALU.
__global__ __launch_bounds__(32) void attn_kernel(
    const bf16* __restrict__ qbf, const bf16* __restrict__ kbf,
    const bf16* __restrict__ vt, const float* __restrict__ rtab,
    const int* __restrict__ rpb, bf16* __restrict__ att) {
  __shared__ float biasT[16 * BSTRIDE];   // [i_local][col 0..576]
  __shared__ bf16  ldsP[16 * 32];

  const int lane = threadIdx.x, ln = lane & 15, lh = lane >> 4;
  const int qt = blockIdx.x;
  const int bh = blockIdx.y, b = bh >> 4, h = bh & 15;
  const int qbase = qt * 16;

  // ---- one-time bias slab: biasT[il][col] = rpe_tab[h][i][rp_bucket[i][col]] ----
  for (int idx = lane; idx < 16 * (Pp + 1); idx += 32) {
    int il = idx / (Pp + 1);
    int col = idx % (Pp + 1);
    int i = qbase + il;
    if (i > LQ - 1) i = LQ - 1;
    int bucket = rpb[i * LQ + col];
    biasT[il * BSTRIDE + col] = rtab[((h * LQ + i) << 3) + bucket];
  }

  // persistent Q A-fragments (d 0..31 and 32..63 of this head)
  const bf16* qrow = qbf + (size_t)(qbase + ln) * Cc + h * HD;
  v16bf a0 = frag2x8(qrow + lh * 16,      qrow + 8 + lh * 16);
  v16bf a1 = frag2x8(qrow + 32 + lh * 16, qrow + 40 + lh * 16);

  v8f zero = {};
  v8f acc[4] = {zero, zero, zero, zero};
  float mrow[8], lrow[8];
  #pragma unroll
  for (int p = 0; p < 8; ++p) { mrow[p] = -3.0e38f; lrow[p] = 0.f; }

  const size_t kbase = (size_t)bh * NP * HD;
  const size_t vbase = (size_t)bh * HD * NP;
  const int NB = NP / 32;

  auto loadK = [&](int jb, int jt, int ks) -> v16bf {
    const bf16* kp = kbf + kbase + (size_t)(jb + jt * 16 + ln) * HD + ks * 32 + lh * 16;
    return frag16c(kp);
  };

  __syncthreads();   // bias slab ready

  // double-buffered K fragments across key blocks
  v16bf kc00 = loadK(0, 0, 0), kc01 = loadK(0, 0, 1);
  v16bf kc10 = loadK(0, 1, 0), kc11 = loadK(0, 1, 1);

  for (int kb = 0; kb < NB; ++kb) {
    const int jb = kb * 32;
    const int jbn = (kb + 1 < NB) ? jb + 32 : jb;    // clamped prefetch target
    if (jb + 64 < NP)   // stream-prefetch block kb+2 (global_prefetch_b8)
      __builtin_prefetch(kbf + kbase + (size_t)(jb + 64) * HD, 0, 1);

    // prefetch next block's K fragments (consumed next iteration)
    v16bf kn00 = loadK(jbn, 0, 0), kn01 = loadK(jbn, 0, 1);
    v16bf kn10 = loadK(jbn, 1, 0), kn11 = loadK(jbn, 1, 1);

    // V fragments for THIS block: issued now, consumed after softmax
    v16bf vf0 = frag16c(vt + vbase + (size_t)(0 * 16 + ln) * NP + jb + lh * 16);
    v16bf vf1 = frag16c(vt + vbase + (size_t)(1 * 16 + ln) * NP + jb + lh * 16);
    v16bf vf2 = frag16c(vt + vbase + (size_t)(2 * 16 + ln) * NP + jb + lh * 16);
    v16bf vf3 = frag16c(vt + vbase + (size_t)(3 * 16 + ln) * NP + jb + lh * 16);

    // S = Q K^T for the two 16-key tiles (K frags were loaded last iteration)
    v8f c0 = wmma_bf16(a1, kc01, wmma_bf16(a0, kc00, zero));
    v8f c1 = wmma_bf16(a1, kc11, wmma_bf16(a0, kc10, zero));

    // scale + bias (LDS slab) + pad mask
    float s[2][8];
    #pragma unroll
    for (int jt = 0; jt < 2; ++jt) {
      const int j = jb + jt * 16 + ln;
      const bool valid = (j < Nn);
      const int col = (j == 0) ? 0 : ((j - 1) % Pp) + 1;   // modality tiling
      const v8f& c = jt ? c1 : c0;
      #pragma unroll
      for (int p = 0; p < 8; ++p) {
        float sv = c[p] * SCALE + biasT[(p + 8 * lh) * BSTRIDE + col];
        s[jt][p] = valid ? sv : -3.0e38f;
      }
    }

    // online softmax: per-row max/sum across each 16-lane half
    #pragma unroll
    for (int p = 0; p < 8; ++p) {
      float rm = fmaxf(s[0][p], s[1][p]);
      for (int mk = 8; mk >= 1; mk >>= 1) rm = fmaxf(rm, __shfl_xor(rm, mk, 32));
      float mn = fmaxf(mrow[p], rm);
      float corr = __expf(mrow[p] - mn);
      float e0 = __expf(s[0][p] - mn);
      float e1 = __expf(s[1][p] - mn);
      float rs = e0 + e1;
      for (int mk = 8; mk >= 1; mk >>= 1) rs += __shfl_xor(rs, mk, 32);
      lrow[p] = lrow[p] * corr + rs;
      mrow[p] = mn;
      #pragma unroll
      for (int c4 = 0; c4 < 4; ++c4) acc[c4][p] *= corr;
      s[0][p] = e0;
      s[1][p] = e1;
    }

    // restripe P (C-layout) -> A-layout through LDS, then O += P*V
    #pragma unroll
    for (int jt = 0; jt < 2; ++jt)
      #pragma unroll
      for (int p = 0; p < 8; ++p)
        ldsP[(p + 8 * lh) * 32 + jt * 16 + ln] = (bf16)s[jt][p];
    __syncthreads();
    {
      const bf16* lp = &ldsP[ln * 32 + lh * 16];
      v16bf pa = frag2x8(lp, lp + 8);
      acc[0] = wmma_bf16(pa, vf0, acc[0]);
      acc[1] = wmma_bf16(pa, vf1, acc[1]);
      acc[2] = wmma_bf16(pa, vf2, acc[2]);
      acc[3] = wmma_bf16(pa, vf3, acc[3]);
    }
    __syncthreads();

    kc00 = kn00; kc01 = kn01; kc10 = kn10; kc11 = kn11;
  }

  // normalize and emit (bf16, padded-row layout for the projection GEMM)
  #pragma unroll
  for (int ct = 0; ct < 4; ++ct)
    #pragma unroll
    for (int p = 0; p < 8; ++p) {
      int i = qbase + p + 8 * lh;
      float o = acc[ct][p] / fmaxf(lrow[p], 1e-30f);
      att[((size_t)(b * LQP + i)) * Cc + h * HD + ct * 16 + ln] = (bf16)o;
    }
}

// ---------------- output projection: out = att @ Wp^T + bp ----------------
// grid = (LQP/16, B, 16); one wave computes 16 rows x 64 cols, double-buffered.
__global__ __launch_bounds__(32) void proj_kernel(
    const bf16* __restrict__ att, const bf16* __restrict__ wpbf,
    const float* __restrict__ bp, float* __restrict__ out) {
  const int lane = threadIdx.x, ln = lane & 15, lh = lane >> 4;
  const int rt = blockIdx.x, b = blockIdx.y, cg = blockIdx.z;

  const bf16* arow = att + (size_t)(b * LQP + rt * 16 + ln) * Cc;
  auto loadA = [&](int ks) -> v16bf {
    const bf16* ap = arow + ks * 32 + lh * 16;
    return frag2x8(ap, ap + 8);
  };
  auto loadB = [&](int ks, int ct) -> v16bf {
    return frag16c(wpbf + (size_t)(cg * 64 + ct * 16 + ln) * Cc + (size_t)ks * 32 + lh * 16);
  };

  v8f zero = {};
  v8f acc[4] = {zero, zero, zero, zero};

  v16bf a_c  = loadA(0);
  v16bf b_c0 = loadB(0, 0), b_c1 = loadB(0, 1), b_c2 = loadB(0, 2), b_c3 = loadB(0, 3);
  for (int ks = 0; ks < Cc / 32 - 1; ++ks) {
    v16bf a_n  = loadA(ks + 1);
    v16bf b_n0 = loadB(ks + 1, 0), b_n1 = loadB(ks + 1, 1);
    v16bf b_n2 = loadB(ks + 1, 2), b_n3 = loadB(ks + 1, 3);
    acc[0] = wmma_bf16(a_c, b_c0, acc[0]);
    acc[1] = wmma_bf16(a_c, b_c1, acc[1]);
    acc[2] = wmma_bf16(a_c, b_c2, acc[2]);
    acc[3] = wmma_bf16(a_c, b_c3, acc[3]);
    a_c = a_n; b_c0 = b_n0; b_c1 = b_n1; b_c2 = b_n2; b_c3 = b_n3;
  }
  acc[0] = wmma_bf16(a_c, b_c0, acc[0]);
  acc[1] = wmma_bf16(a_c, b_c1, acc[1]);
  acc[2] = wmma_bf16(a_c, b_c2, acc[2]);
  acc[3] = wmma_bf16(a_c, b_c3, acc[3]);

  #pragma unroll
  for (int ct = 0; ct < 4; ++ct)
    #pragma unroll
    for (int p = 0; p < 8; ++p) {
      int i = rt * 16 + p + 8 * lh;
      if (i < LQ) {
        int c = cg * 64 + ct * 16 + ln;
        out[((size_t)(b * LQ + i)) * Cc + c] = acc[ct][p] + bp[c];
      }
    }
}

extern "C" void kernel_launch(void* const* d_in, const int* in_sizes, int n_in,
                              void* d_out, int out_size, void* d_ws, size_t ws_size,
                              hipStream_t stream) {
  (void)n_in; (void)out_size; (void)ws_size;
  const float* x    = (const float*)d_in[0];
  const float* ql   = (const float*)d_in[1];
  const float* pe   = (const float*)d_in[2];
  const float* Wk   = (const float*)d_in[3];
  const float* Wv   = (const float*)d_in[4];
  const float* Wp   = (const float*)d_in[5];
  const float* bp   = (const float*)d_in[6];
  const float* rpeW = (const float*)d_in[7];
  const int*   rpb  = (const int*)d_in[8];
  const int nb = in_sizes[7] / HD;   // number of RPE buckets (5 for this config)

  char* ws = (char*)d_ws;
  size_t off = 0;
  auto take = [&](size_t bytes) -> void* {
    void* p = ws + off;
    off = (off + bytes + 255) & ~(size_t)255;
    return p;
  };
  bf16*  xbf  = (bf16*)take((size_t)Bb * Nn * Cc * 2);
  bf16*  wkbf = (bf16*)take((size_t)Cc * Cc * 2);
  bf16*  wvbf = (bf16*)take((size_t)Cc * Cc * 2);
  bf16*  wpbf = (bf16*)take((size_t)Cc * Cc * 2);
  bf16*  qbf  = (bf16*)take((size_t)LQP * Cc * 2);
  float* rtab = (float*)take((size_t)Hh * LQ * 8 * 4);
  bf16*  kbf  = (bf16*)take((size_t)Bb * Hh * NP * HD * 2);
  bf16*  vtb  = (bf16*)take((size_t)Bb * Hh * HD * NP * 2);
  bf16*  attb = (bf16*)take((size_t)Bb * LQP * Cc * 2);

  // 1) fp32 -> bf16 conversions
  cvt_kernel<<<dim3(2048), dim3(256), 0, stream>>>(x, xbf, Bb * Nn * Cc);
  cvt_kernel<<<dim3(1024), dim3(256), 0, stream>>>(Wk, wkbf, Cc * Cc);
  cvt_kernel<<<dim3(1024), dim3(256), 0, stream>>>(Wv, wvbf, Cc * Cc);
  cvt_kernel<<<dim3(1024), dim3(256), 0, stream>>>(Wp, wpbf, Cc * Cc);

  // 2) q (batch-independent) + RPE table
  qprep_kernel<<<dim3((LQP * Cc + 255) / 256), dim3(256), 0, stream>>>(ql, pe, qbf);
  rpe_kernel<<<dim3((Hh * LQ * nb + 255) / 256), dim3(256), 0, stream>>>(ql, pe, rpeW,
                                                                         rtab, nb);
  // 3) K / V projections into attention-friendly layouts
  kv_gemm_kernel<<<dim3(NP / 16, Bb * Hh, 2), dim3(32), 0, stream>>>(xbf, wkbf, wvbf,
                                                                     kbf, vtb);
  // 4) fused flash attention with contextual RPE bias
  attn_kernel<<<dim3(LQP / 16, Bb * Hh), dim3(32), 0, stream>>>(qbf, kbf, vtb, rtab,
                                                                rpb, attb);
  // 5) output projection + bias
  proj_kernel<<<dim3(LQP / 16, Bb, Cc / HD), dim3(32), 0, stream>>>(attb, wpbf, bp,
                                                                    (float*)d_out);
}
